// TranscendentPerfectionPiMoE_2027224564571
// MI455X (gfx1250) — matrix operand order repaired and checked by reading, not compiled, not run on hardware
//
#include <hip/hip_runtime.h>
#include <cstdint>

// ---------------------------------------------------------------- constants
constexpr int E  = 4;
constexpr int B  = 8;
constexpr int S  = 128;
constexpr int D  = 512;
constexpr int P  = 1024;
constexpr int H  = 16;
constexpr int DH = 64;          // P / H
constexpr int BS = B * S;       // 1024 tokens
constexpr int P4 = 4 * P;       // 4096

typedef __bf16 bf16_t;
typedef __attribute__((ext_vector_type(16))) __bf16 v16bf;
typedef __attribute__((ext_vector_type(8)))  __bf16 v8bf;
typedef __attribute__((ext_vector_type(8)))  float  v8f;

// ---------------------------------------------------------------- helpers
static __device__ inline bf16_t f2bf(float f) {
    union { float f; unsigned u; } a; a.f = f;
    unsigned u = a.u + 0x7fffu + ((a.u >> 16) & 1u);   // round-nearest-even
    union { unsigned short s; bf16_t b; } r; r.s = (unsigned short)(u >> 16);
    return r.b;
}
static __device__ inline float sigm(float x) { return 1.f / (1.f + expf(-x)); }

// v_wmma_f32_16x16x32_bf16 wrapper (wave32; D = A*B + C, fp32 accum)
static __device__ inline v8f wmma_bf16(v16bf a, v16bf b, v8f c) {
    return __builtin_amdgcn_wmma_f32_16x16x32_bf16(
        /*neg_a=*/false, a, /*neg_b=*/false, b,
        /*c_mod=*/(short)0, c, /*reuse_a=*/false, /*reuse_b=*/false);
}

// gfx1250 async copy: 16B global -> LDS, bypassing VGPRs (tracked by ASYNCcnt).
// LDS address is taken from the vdst VGPR; generic-LDS address low 32 bits == LDS offset.
static __device__ inline void async_copy16(void* lds, const void* gptr) {
    unsigned lds_off = (unsigned)(size_t)lds;
    unsigned long long ga = (unsigned long long)(size_t)gptr;
    asm volatile("global_load_async_to_lds_b128 %0, %1, off"
                 :: "v"(lds_off), "v"(ga) : "memory");
}
static __device__ inline void async_wait0() {
    asm volatile("s_wait_asynccnt 0x0" ::: "memory");
}

// A-fragment (16x32 bf16): lane<16 holds M=lane, K in {0..7,16..23};
// lane>=16 holds M=lane-16, K in {8..15,24..31}. `row` points at As[m][0](+kc).
static __device__ inline v16bf ld_fragA(const bf16_t* row, int lane) {
    int kA = (lane >> 4) * 8;
    v8bf lo = *(const v8bf*)(row + kA);
    v8bf hi = *(const v8bf*)(row + kA + 16);
    return __builtin_shufflevector(lo, hi, 0,1,2,3,4,5,6,7,8,9,10,11,12,13,14,15);
}
// B-fragment (32x16 bf16, staged transposed Bt[n][k]): lane<16 -> N=lane,
// K in {0..7,16..23}; lane>=16 -> N=lane-16, K in {8..15,24..31}.
static __device__ inline v16bf ld_fragB(const bf16_t* row, int lane) {
    int kB = (lane >> 4) * 8;
    v8bf lo = *(const v8bf*)(row + kB);
    v8bf hi = *(const v8bf*)(row + kB + 16);
    return __builtin_shufflevector(lo, hi, 0,1,2,3,4,5,6,7,8,9,10,11,12,13,14,15);
}

// ---------------------------------------------------------------- generic GEMM
// C[M,N] = A[M,K](bf16) * W[e][K,N](fp32->bf16 in LDS) + bias, fused epilogue.
// EPI: 0=f32+bias  1=bf16 relu  2=bf16 bias  3=f32 sigmoid  4=bf16 tanh*perf
template<int EPI>
__global__ __launch_bounds__(256) void gemm_bf16_kernel(
    const bf16_t* __restrict__ A, long aEStride,
    const float*  __restrict__ Bw,
    const float*  __restrict__ bias,
    void*         __restrict__ Out, long outEStride,
    const float*  __restrict__ perf, long perfEStride,
    int M, int N, int K)
{
    int e = blockIdx.z;
    const bf16_t* Ae = A  + (long)e * aEStride;
    const float*  Be = Bw + (long)e * (long)K * N;
    const float*  be = bias + (long)e * N;

    int m0 = blockIdx.y * 64, n0 = blockIdx.x * 64;
    __shared__ alignas(16) bf16_t As[64][32];
    __shared__ alignas(16) bf16_t Bt[64][32];   // transposed: [n][k]

    int tid = threadIdx.x, w = tid >> 5, l = tid & 31;
    int r = w >> 1, cbase = (w & 1) * 32;       // 4x2 waves -> 64x64 C tile

    v8f z = {0.f,0.f,0.f,0.f,0.f,0.f,0.f,0.f};
    v8f acc0 = z, acc1 = z;

    for (int k0 = 0; k0 < K; k0 += 32) {
        // stage A chunk 64x32: async 16B copies straight into LDS (no VGPRs)
        { int idx = tid * 8; int m = idx >> 5; int kk = idx & 31;
          async_copy16(&As[m][kk], Ae + (long)(m0 + m) * K + (k0 + kk)); }
        if (k0 + 32 < K)   // gfx1250 global_prefetch_b8 on next weight chunk
            __builtin_prefetch(Be + (long)(k0 + 32) * N + n0 + (tid & 63), 0, 1);
        // stage W chunk 32x64 fp32 -> bf16, transposed into Bt[n][k]
        #pragma unroll
        for (int i = 0; i < 8; i++) {
            int id = tid + (i << 8); int k = id >> 6, nn = id & 63;
            Bt[nn][k] = f2bf(Be[(long)(k0 + k) * N + n0 + nn]);
        }
        async_wait0();
        __syncthreads();
        int mr = r * 16 + (l & 15);
        v16bf a  = ld_fragA(&As[mr][0], l);
        v16bf b0 = ld_fragB(&Bt[cbase + (l & 15)][0], l);
        v16bf b1 = ld_fragB(&Bt[cbase + 16 + (l & 15)][0], l);
        acc0 = wmma_bf16(a, b0, acc0);
        acc1 = wmma_bf16(a, b1, acc1);
        __syncthreads();
    }

    int col = l & 15, rb = (l >> 4) * 8;
    long gm0 = m0 + r * 16 + rb;
    int  gn0 = n0 + cbase + col, gn1 = gn0 + 16;
    float bb0 = be[gn0], bb1 = be[gn1];
    #pragma unroll
    for (int j = 0; j < 8; j++) {
        long gm = gm0 + j;
        float v0 = acc0[j] + bb0, v1 = acc1[j] + bb1;
        long o0 = gm * (long)N + gn0, o1 = gm * (long)N + gn1;
        if (EPI == 0) { float* O = (float*)Out + (long)e * outEStride;
                        O[o0] = v0; O[o1] = v1; }
        if (EPI == 1) { bf16_t* O = (bf16_t*)Out + (long)e * outEStride;
                        O[o0] = f2bf(fmaxf(v0, 0.f)); O[o1] = f2bf(fmaxf(v1, 0.f)); }
        if (EPI == 2) { bf16_t* O = (bf16_t*)Out + (long)e * outEStride;
                        O[o0] = f2bf(v0); O[o1] = f2bf(v1); }
        if (EPI == 3) { float* O = (float*)Out + (long)e * outEStride;
                        O[o0] = sigm(v0); O[o1] = sigm(v1); }
        if (EPI == 4) { const float* Pf = perf + (long)e * perfEStride;
                        bf16_t* O = (bf16_t*)Out + (long)e * outEStride;
                        O[o0] = f2bf(tanhf(v0) * Pf[o0]);
                        O[o1] = f2bf(tanhf(v1) * Pf[o1]); }
    }
}

// ---------------------------------------------------------------- attention
// one block per (e,b,h): scores = q k^T / 8, softmax, out = p v  (all WMMA)
__global__ __launch_bounds__(256) void attn_kernel(
    const bf16_t* __restrict__ Qb, const bf16_t* __restrict__ Kb,
    const bf16_t* __restrict__ Vb, bf16_t* __restrict__ Ob)
{
    int h = blockIdx.x, b = blockIdx.y, e = blockIdx.z;
    extern __shared__ char smem[];
    bf16_t* qv = (bf16_t*)smem;                      // q [S][DH]; later v^T [DH][S]
    bf16_t* ks = (bf16_t*)(smem + 16384);            // k [S][DH]
    float*  sc = (float*)(smem + 32768);             // scores [S][S] fp32
    bf16_t* pr = (bf16_t*)(smem + 32768 + 65536);    // probs  [S][S] bf16

    int tid = threadIdx.x, w = tid >> 5, l = tid & 31;
    long base = ((long)e * BS + (long)b * S) * P + (long)h * DH;

    #pragma unroll
    for (int j = 0; j < 4; j++) {                    // stage q,k (async 16B chunks)
        int vid = tid + j * 256; int s = vid >> 3; int d = (vid & 7) * 8;
        async_copy16(&qv[s * DH + d], &Qb[base + (long)s * P + d]);
        async_copy16(&ks[s * DH + d], &Kb[base + (long)s * P + d]);
    }
    async_wait0();
    __syncthreads();

    v8f z = {0.f,0.f,0.f,0.f,0.f,0.f,0.f,0.f};
    v8f accs[8];
    #pragma unroll
    for (int c = 0; c < 8; c++) accs[c] = z;
    #pragma unroll
    for (int kc = 0; kc < DH; kc += 32) {            // scores GEMM, K=64
        v16bf a = ld_fragA(&qv[(w * 16 + (l & 15)) * DH + kc], l);
        #pragma unroll
        for (int c = 0; c < 8; c++) {
            v16bf bb = ld_fragB(&ks[(c * 16 + (l & 15)) * DH + kc], l);
            accs[c] = wmma_bf16(a, bb, accs[c]);
        }
    }
    int col = l & 15, rb = (l >> 4) * 8;
    #pragma unroll
    for (int c = 0; c < 8; c++)
        #pragma unroll
        for (int j = 0; j < 8; j++)
            sc[(w * 16 + rb + j) * S + c * 16 + col] = accs[c][j] * 0.125f;
    __syncthreads();

    for (int i = 0; i < 32; i++) {                   // stage v^T into qv region
        int id = tid + i * 256; int s = id >> 6; int d = id & 63;
        qv[d * S + s] = Vb[base + (long)s * P + d];
    }
    if (tid < S) {                                   // row softmax
        float mx = -1e30f;
        for (int j = 0; j < S; j++) mx = fmaxf(mx, sc[tid * S + j]);
        float sum = 0.f;
        for (int j = 0; j < S; j++) sum += expf(sc[tid * S + j] - mx);
        float inv = 1.f / sum;
        for (int j = 0; j < S; j++) pr[tid * S + j] = f2bf(expf(sc[tid * S + j] - mx) * inv);
    }
    __syncthreads();

    v8f acco[4];
    #pragma unroll
    for (int c = 0; c < 4; c++) acco[c] = z;
    #pragma unroll
    for (int kc = 0; kc < S; kc += 32) {             // out GEMM, K=128
        v16bf a = ld_fragA(&pr[(w * 16 + (l & 15)) * S + kc], l);
        #pragma unroll
        for (int c = 0; c < 4; c++) {
            v16bf bb = ld_fragB(&qv[(c * 16 + (l & 15)) * S + kc], l);
            acco[c] = wmma_bf16(a, bb, acco[c]);
        }
    }
    #pragma unroll
    for (int c = 0; c < 4; c++)
        #pragma unroll
        for (int j = 0; j < 8; j++)
            Ob[base + (long)(w * 16 + rb + j) * P + c * 16 + col] = f2bf(acco[c][j]);
}

// ---------------------------------------------------------------- LSTM step
// block = (16 units, expert): 4 waves each do one 16x16 gate tile of
// h_prev[8(pad16),P] @ lWh[:, g*P+u0 : +16], fused i/f/g/o cell epilogue.
__global__ __launch_bounds__(128) void lstm_step_kernel(
    const float* __restrict__ lWh, const float* __restrict__ xpart,
    const float* __restrict__ hsIn, float* __restrict__ hsOut,
    float* __restrict__ cs, bf16_t* __restrict__ Mb, int t)
{
    int u0 = blockIdx.x * 16, e = blockIdx.y;
    const float* Wh  = lWh  + (long)e * P * P4;
    const float* hIn = hsIn + (long)e * B * P;
    float* hOut = hsOut + (long)e * B * P;
    float* cse  = cs    + (long)e * B * P;

    __shared__ alignas(16) bf16_t As[16][32];
    __shared__ alignas(16) bf16_t Bt[4][16][32];
    __shared__ float gb[4][8][16];

    int tid = threadIdx.x, w = tid >> 5, l = tid & 31;
    v8f acc = {0.f,0.f,0.f,0.f,0.f,0.f,0.f,0.f};

    for (int k0 = 0; k0 < P; k0 += 32) {
        #pragma unroll
        for (int i = 0; i < 4; i++) {                // h_prev -> bf16 (rows 8..15 = 0)
            int id = tid * 4 + i; int m = id >> 5; int kk = id & 31;
            As[m][kk] = (m < 8) ? f2bf(hIn[m * P + k0 + kk]) : f2bf(0.f);
        }
        #pragma unroll
        for (int i = 0; i < 16; i++) {               // 4 gate weight tiles, transposed
            int id = tid + (i << 7); int g = id >> 9; int rem = id & 511;
            int k = rem >> 4; int nn = rem & 15;
            Bt[g][nn][k] = f2bf(Wh[(long)(k0 + k) * P4 + (long)g * P + u0 + nn]);
        }
        __syncthreads();
        v16bf a  = ld_fragA(&As[l & 15][0], l);
        v16bf bb = ld_fragB(&Bt[w][l & 15][0], l);
        acc = wmma_bf16(a, bb, acc);
        __syncthreads();
    }
    if (l < 16) {                                    // rows 0..7 = valid batches
        #pragma unroll
        for (int j = 0; j < 8; j++) gb[w][j][l] = acc[j];
    }
    __syncthreads();
    {
        int b = tid >> 4, nn = tid & 15; int u = u0 + nn;
        long xrow = ((long)e * BS + (long)b * S + t) * P4;
        float gi = gb[0][b][nn] + xpart[xrow + u];
        float gf = gb[1][b][nn] + xpart[xrow + P + u];
        float gg = gb[2][b][nn] + xpart[xrow + 2 * P + u];
        float go = gb[3][b][nn] + xpart[xrow + 3 * P + u];
        float c  = sigm(gf) * cse[b * P + u] + sigm(gi) * tanhf(gg);
        float hn = sigm(go) * tanhf(c);
        cse[b * P + u] = c;
        hOut[b * P + u] = hn;
        Mb[((long)e * BS + (long)b * S + t) * P + u] = f2bf(hn);
    }
}

// ---------------------------------------------------------------- small kernels
__global__ void ln_rows_kernel(const float* __restrict__ X,
                               const float* __restrict__ gamma,
                               const float* __restrict__ beta,
                               void* __restrict__ out, int W, int outBf16)
{
    long row = blockIdx.x; int e = (int)(row / BS);
    const float* xr = X + row * (long)W;
    int tid = threadIdx.x, n = W >> 7;
    float v[8]; float s = 0.f;
    for (int i = 0; i < n; i++) { v[i] = xr[tid + (i << 7)]; s += v[i]; }
    __shared__ float red[128];
    red[tid] = s; __syncthreads();
    for (int o = 64; o > 0; o >>= 1) { if (tid < o) red[tid] += red[tid + o]; __syncthreads(); }
    float mean = red[0] / W;
    __syncthreads();
    float q = 0.f;
    for (int i = 0; i < n; i++) { float d = v[i] - mean; q += d * d; }
    red[tid] = q; __syncthreads();
    for (int o = 64; o > 0; o >>= 1) { if (tid < o) red[tid] += red[tid + o]; __syncthreads(); }
    float rinv = rsqrtf(red[0] / W + 1e-5f);
    for (int i = 0; i < n; i++) {
        int idx = tid + (i << 7);
        long gi = (long)e * W + idx;
        float val = (v[i] - mean) * rinv * gamma[gi] + beta[gi];
        if (outBf16) ((bf16_t*)out)[row * (long)W + idx] = f2bf(val);
        else         ((float*)out)[row * (long)W + idx]  = val;
    }
}

__global__ void gating_kernel(const float* __restrict__ x, const float* __restrict__ Wg,
                              const float* __restrict__ bg, float* __restrict__ gates)
{
    __shared__ float means[B][D];
    __shared__ float lg[B * E];
    int tid = threadIdx.x;
    if (tid < D) {
        for (int b = 0; b < B; b++) {
            float s = 0.f;
            for (int ss = 0; ss < S; ss++) s += x[((long)b * S + ss) * D + tid];
            means[b][tid] = s * (1.f / S);
        }
    }
    __syncthreads();
    if (tid < B * E) {
        int b = tid >> 2, e2 = tid & 3;
        float a = bg[e2];
        for (int d = 0; d < D; d++) a += means[b][d] * Wg[d * E + e2];
        lg[tid] = a;
    }
    __syncthreads();
    if (tid < B) {
        const float* l4 = &lg[tid * E];
        int i1 = 0;
        for (int e2 = 1; e2 < E; e2++) if (l4[e2] > l4[i1]) i1 = e2;
        int i2 = -1;
        for (int e2 = 0; e2 < E; e2++)
            if (e2 != i1 && (i2 < 0 || l4[e2] > l4[i2])) i2 = e2;
        float ex = expf(l4[i2] - l4[i1]);
        float sum = 1.f + ex;
        for (int e2 = 0; e2 < E; e2++) gates[tid * E + e2] = 0.f;
        gates[tid * E + i1] = 1.f / sum;
        gates[tid * E + i2] = ex / sum;
    }
}

__global__ void combine_kernel(const float* __restrict__ gates,
                               const float* __restrict__ EOb, float* __restrict__ out)
{
    long idx = (long)blockIdx.x * 256 + threadIdx.x;   // < BS*D
    int b = (int)(idx / ((long)S * D));
    float a = 0.f;
    #pragma unroll
    for (int e2 = 0; e2 < E; e2++)
        a += gates[b * E + e2] * EOb[(long)e2 * BS * D + idx];
    out[idx] = a;
}

__global__ void cvt_bf16_kernel(const float* __restrict__ in, bf16_t* __restrict__ out, long n)
{ long i = (long)blockIdx.x * 256 + threadIdx.x; if (i < n) out[i] = f2bf(in[i]); }

__global__ void zero_f32_kernel(float* p, long n)
{ long i = (long)blockIdx.x * 256 + threadIdx.x; if (i < n) p[i] = 0.f; }

// ---------------------------------------------------------------- workspace map
constexpr size_t MB1 = 1ull << 20;
constexpr size_t OFF_XB  = 0;                    // bf16 [BS][D]        1 MB
constexpr size_t OFF_A1  = OFF_XB  + 1  * MB1;   // bf16 [E][BS][P]     8 MB
constexpr size_t OFF_A2  = OFF_A1  + 8  * MB1;   // bf16 [E][BS][P]     8 MB
constexpr size_t OFF_QB  = OFF_A2  + 8  * MB1;   // bf16 [E][BS][P]     8 MB
constexpr size_t OFF_KB  = OFF_QB  + 8  * MB1;
constexpr size_t OFF_VB  = OFF_KB  + 8  * MB1;
constexpr size_t OFF_MB  = OFF_VB  + 8  * MB1;   // bf16 [E][BS][P]  (LSTM outputs)
constexpr size_t OFF_F0  = OFF_MB  + 8  * MB1;   // f32  [E][BS][4P]   64 MB (xpart)
constexpr size_t OFF_F1  = OFF_F0  + 64 * MB1;   // f32  [E][BS][P]    16 MB
constexpr size_t OFF_EO  = OFF_F1  + 16 * MB1;   // f32  [E][BS][D]     8 MB
constexpr size_t OFF_HS0 = OFF_EO  + 8  * MB1;   // f32  [E][B][P]    128 KB
constexpr size_t OFF_HS1 = OFF_HS0 + 131072;
constexpr size_t OFF_CS  = OFF_HS1 + 131072;
constexpr size_t OFF_GT  = OFF_CS  + 131072;     // f32 [B][E]

// ---------------------------------------------------------------- launcher
extern "C" void kernel_launch(void* const* d_in, const int* in_sizes, int n_in,
                              void* d_out, int out_size, void* d_ws, size_t ws_size,
                              hipStream_t stream)
{
    (void)in_sizes; (void)n_in; (void)out_size; (void)ws_size;
    const float* x   = (const float*)d_in[0];
    const float* Wg  = (const float*)d_in[1];
    const float* bg  = (const float*)d_in[2];
    const float* eW1 = (const float*)d_in[3];
    const float* eb1 = (const float*)d_in[4];
    const float* eW2 = (const float*)d_in[5];
    const float* eb2 = (const float*)d_in[6];
    const float* eg  = (const float*)d_in[7];
    const float* ebt = (const float*)d_in[8];
    const float* Wq  = (const float*)d_in[9];
    const float* Wk  = (const float*)d_in[10];
    const float* Wv  = (const float*)d_in[11];
    const float* Wo  = (const float*)d_in[12];
    const float* bq  = (const float*)d_in[13];
    const float* bk  = (const float*)d_in[14];
    const float* bv  = (const float*)d_in[15];
    const float* bo  = (const float*)d_in[16];
    const float* lWx = (const float*)d_in[17];
    const float* lWh = (const float*)d_in[18];
    const float* lb  = (const float*)d_in[19];
    const float* pW1 = (const float*)d_in[20];
    const float* pb1 = (const float*)d_in[21];
    const float* pW2 = (const float*)d_in[22];
    const float* pb2 = (const float*)d_in[23];
    const float* xW1 = (const float*)d_in[24];
    const float* xb1 = (const float*)d_in[25];
    const float* xW2 = (const float*)d_in[26];
    const float* xb2 = (const float*)d_in[27];
    const float* dW1 = (const float*)d_in[28];
    const float* db1 = (const float*)d_in[29];
    const float* dW2 = (const float*)d_in[30];
    const float* db2 = (const float*)d_in[31];
    const float* dgv = (const float*)d_in[32];
    const float* dbt = (const float*)d_in[33];

    char* ws = (char*)d_ws;
    bf16_t* XB = (bf16_t*)(ws + OFF_XB);
    bf16_t* A1 = (bf16_t*)(ws + OFF_A1);
    bf16_t* A2 = (bf16_t*)(ws + OFF_A2);
    bf16_t* QB = (bf16_t*)(ws + OFF_QB);
    bf16_t* KB = (bf16_t*)(ws + OFF_KB);
    bf16_t* VB = (bf16_t*)(ws + OFF_VB);
    bf16_t* Mb = (bf16_t*)(ws + OFF_MB);
    float*  F0 = (float*) (ws + OFF_F0);
    float*  F1 = (float*) (ws + OFF_F1);
    float*  EO = (float*) (ws + OFF_EO);
    float*  HS0 = (float*)(ws + OFF_HS0);
    float*  HS1 = (float*)(ws + OFF_HS1);
    float*  CS  = (float*)(ws + OFF_CS);
    float*  GT  = (float*)(ws + OFF_GT);

    const long actP = (long)BS * P;   // per-expert activation stride (P wide)

    // gating + x -> bf16
    gating_kernel<<<1, 512, 0, stream>>>(x, Wg, bg, GT);
    cvt_bf16_kernel<<<(BS * D + 255) / 256, 256, 0, stream>>>(x, XB, (long)BS * D);

    dim3 gP(P / 64, BS / 64, E);
    // encoder
    gemm_bf16_kernel<1><<<gP, 256, 0, stream>>>(XB, 0L, eW1, eb1, A1, actP, nullptr, 0L, BS, P, D);
    gemm_bf16_kernel<0><<<gP, 256, 0, stream>>>(A1, actP, eW2, eb2, F1, actP, nullptr, 0L, BS, P, P);
    ln_rows_kernel<<<E * BS, 128, 0, stream>>>(F1, eg, ebt, A2, P, 1);
    // qkv projections
    gemm_bf16_kernel<2><<<gP, 256, 0, stream>>>(A2, actP, Wq, bq, QB, actP, nullptr, 0L, BS, P, P);
    gemm_bf16_kernel<2><<<gP, 256, 0, stream>>>(A2, actP, Wk, bk, KB, actP, nullptr, 0L, BS, P, P);
    gemm_bf16_kernel<2><<<gP, 256, 0, stream>>>(A2, actP, Wv, bv, VB, actP, nullptr, 0L, BS, P, P);
    // attention (dynamic LDS: q/vT 16K + k 16K + scores 64K + probs 32K)
    attn_kernel<<<dim3(H, B, E), 256, 131072, stream>>>(QB, KB, VB, A1);
    // output projection + LSTM x-part
    gemm_bf16_kernel<2><<<gP, 256, 0, stream>>>(A1, actP, Wo, bo, A2, actP, nullptr, 0L, BS, P, P);
    gemm_bf16_kernel<0><<<dim3(P4 / 64, BS / 64, E), 256, 0, stream>>>(
        A2, actP, lWx, lb, F0, (long)BS * P4, nullptr, 0L, BS, P4, P);
    // LSTM recurrence
    zero_f32_kernel<<<(E * B * P + 255) / 256, 256, 0, stream>>>(HS0, (long)E * B * P);
    zero_f32_kernel<<<(E * B * P + 255) / 256, 256, 0, stream>>>(CS,  (long)E * B * P);
    float* hIn = HS0; float* hOut = HS1;
    for (int t = 0; t < S; t++) {
        lstm_step_kernel<<<dim3(P / 16, E), 128, 0, stream>>>(lWh, F0, hIn, hOut, CS, Mb, t);
        float* tmp = hIn; hIn = hOut; hOut = tmp;
    }
    // perf branch (sigmoid)
    gemm_bf16_kernel<1><<<gP, 256, 0, stream>>>(Mb, actP, pW1, pb1, A1, actP, nullptr, 0L, BS, P, P);
    gemm_bf16_kernel<3><<<gP, 256, 0, stream>>>(A1, actP, pW2, pb2, F1, actP, nullptr, 0L, BS, P, P);
    // exc branch (tanh) * perf -> o2 (bf16)
    gemm_bf16_kernel<1><<<gP, 256, 0, stream>>>(Mb, actP, xW1, xb1, A2, actP, nullptr, 0L, BS, P, P);
    gemm_bf16_kernel<4><<<gP, 256, 0, stream>>>(A2, actP, xW2, xb2, A1, actP, F1, actP, BS, P, P);
    // decoder
    gemm_bf16_kernel<1><<<gP, 256, 0, stream>>>(A1, actP, dW1, db1, A2, actP, nullptr, 0L, BS, P, P);
    gemm_bf16_kernel<0><<<dim3(D / 64, BS / 64, E), 256, 0, stream>>>(
        A2, actP, dW2, db2, F1, (long)BS * D, nullptr, 0L, BS, D, P);
    ln_rows_kernel<<<E * BS, 128, 0, stream>>>(F1, dgv, dbt, EO, D, 0);
    // top-2 gated combine
    combine_kernel<<<(BS * D) / 256, 256, 0, stream>>>(GT, EO, (float*)d_out);
}
